// ScaledDotAttention_75179107549594
// MI455X (gfx1250) — matrix operand order repaired and checked by reading, not compiled, not run on hardware
//
#include <hip/hip_runtime.h>
#include <stdint.h>

// ---------------- problem constants ----------------
#define Bc  8
#define Nc  4096
#define DKc 512
#define DMc 64
#define KT  64      // keys per flash tile
#define VSTR 72     // padded LDS row stride (halves): 64 + 8, keeps b128 16B-aligned
#define PSTR 72

typedef __bf16 bf16_t;
typedef float        v8f   __attribute__((ext_vector_type(8)));
typedef bf16_t       v16bf __attribute__((ext_vector_type(16)));
typedef float        f32x4 __attribute__((ext_vector_type(4)));
typedef unsigned int u32;
typedef u32          u32x4 __attribute__((ext_vector_type(4)));
typedef int          i32x4 __attribute__((ext_vector_type(4)));
typedef int          i32x8 __attribute__((ext_vector_type(8)));

union ABv { u32x4 u[2]; v16bf v; };

#define WMMA_BF16(a, b, c) \
  __builtin_amdgcn_wmma_f32_16x16x32_bf16(false, (a), false, (b), (short)0, (c), false, false)

#if defined(__has_builtin)
#if __has_builtin(__builtin_amdgcn_tensor_load_to_lds) && __has_builtin(__builtin_amdgcn_s_wait_tensorcnt)
#define HAVE_TDM 1
#endif
#endif

// ---------------- TDM 2D tile load (D# per CDNA5 ISA) ----------------
#if HAVE_TDM
__device__ __forceinline__ void tdm_load_2d(u32 lds_off, const void* gptr,
                                            u32 td0, u32 td1, u32 stride0,
                                            u32 tile0, u32 tile1) {
  uint64_t ga = (uint64_t)(uintptr_t)gptr;
  u32x4 g0;
  g0[0] = 1u;                                            // count=1, user descriptor
  g0[1] = lds_off;                                       // lds_addr (bytes)
  g0[2] = (u32)ga;                                       // global_addr[31:0]
  g0[3] = (u32)((ga >> 32) & 0x01FFFFFFu) | (2u << 30);  // global_addr[56:32], type=2
  i32x8 g1;
  g1[0] = (int)((1u << 16) |   // data_size = 2 bytes (bf16)
                (1u << 20) |   // pad_enable
                (4u << 22) |   // pad_interval: every 128B (one 64-half row)
                (3u << 25));   // pad_amount: 4 dwords = 16B  -> 72-half LDS stride
  g1[1] = (int)((td0 & 0xffffu) << 16);                  // tensor_dim0[15:0]
  g1[2] = (int)((td0 >> 16) | ((td1 & 0xffffu) << 16));  // tensor_dim0[31:16] | tensor_dim1[15:0]
  g1[3] = (int)((td1 >> 16) | (tile0 << 16));            // tensor_dim1[31:16] | tile_dim0
  g1[4] = (int)(tile1 & 0xffffu);                        // tile_dim1 (tile_dim2 = 0)
  g1[5] = (int)stride0;                                  // tensor_dim0_stride[31:0]
  g1[6] = 0;
  g1[7] = 0;
  i32x4 z4 = {0, 0, 0, 0};
  i32x8 z8 = {0, 0, 0, 0, 0, 0, 0, 0};
  // clang-23 / therock arity: (g0, g1, g2, g3, g4, cpol)
  __builtin_amdgcn_tensor_load_to_lds(g0, g1, z4, z4, z8, 0);
}
#endif

// Fallback cooperative LDS staging (used only if TDM builtin is absent)
__device__ __forceinline__ void copy_tile_fallback(bf16_t* dst, const bf16_t* src,
                                                   int rows, int chunks, size_t sstride,
                                                   int tid, int nthreads) {
  const int total = rows * chunks;
  for (int i = tid; i < total; i += nthreads) {
    const int r = i / chunks, c = i - r * chunks;
    *(u32x4*)(dst + (size_t)r * VSTR + c * 8) = *(const u32x4*)(src + (size_t)r * sstride + c * 8);
  }
}

// ---------------- kernel 0: f32 -> bf16 weight convert ----------------
__global__ void cvt_w_kernel(const float* __restrict__ w, bf16_t* __restrict__ o, int n) {
  const int i = blockIdx.x * 256 + threadIdx.x;
  if (i < n) o[i] = (bf16_t)w[i];
}

// ---------------- kernel 1: projection  out[row][dm] = X[row]·W[dm] + b[dm] ----------------
// A = W (M=dm, 16x32 bf16 fragments straight from bf16 W), B = X^T (K contiguous in X rows).
__global__ void __launch_bounds__(256)
proj_kernel(const float* __restrict__ X, const bf16_t* __restrict__ Wb,
            const float* __restrict__ bias, bf16_t* __restrict__ outp) {
  const int tid  = threadIdx.x;
  const int wave = tid >> 5, lane = tid & 31;
  const int hi   = lane >> 4, ml = lane & 15;
  const int row0 = blockIdx.x * 128 + wave * 16;

  v8f vzero = {};
  v8f acc[4];
#pragma unroll
  for (int mt = 0; mt < 4; ++mt) acc[mt] = vzero;

  for (int c = 0; c < 16; ++c) {     // K = 512 in 16 chunks of 32
    const int kb = c * 32;
    // B fragment: n = ml (X row), k = kb + hi*16 + 0..15, f32 -> bf16 in-register
    const float* xr = X + (size_t)(row0 + ml) * DKc + kb + hi * 16;
    f32x4 x0 = *(const f32x4*)(xr);
    f32x4 x1 = *(const f32x4*)(xr + 4);
    f32x4 x2 = *(const f32x4*)(xr + 8);
    f32x4 x3 = *(const f32x4*)(xr + 12);
    v16bf bv;
#pragma unroll
    for (int j = 0; j < 4; ++j) {
      bv[j]      = (bf16_t)x0[j];
      bv[4 + j]  = (bf16_t)x1[j];
      bv[8 + j]  = (bf16_t)x2[j];
      bv[12 + j] = (bf16_t)x3[j];
    }
#pragma unroll
    for (int mt = 0; mt < 4; ++mt) {
      const bf16_t* wr = Wb + (size_t)(mt * 16 + ml) * DKc + kb + hi * 16;
      ABv wa;
      wa.u[0] = *(const u32x4*)(wr);        // k = kb+hi*16  .. +7
      wa.u[1] = *(const u32x4*)(wr + 8);    // k = kb+hi*16+8.. +15
      acc[mt] = WMMA_BF16(wa.v, bv, acc[mt]);
    }
  }
#pragma unroll
  for (int mt = 0; mt < 4; ++mt)
#pragma unroll
    for (int r = 0; r < 8; ++r) {
      const int dm = mt * 16 + r + hi * 8;          // C: M = dm, N = X row (ml)
      outp[(size_t)(row0 + ml) * DMc + dm] = (bf16_t)(acc[mt][r] + bias[dm]);
    }
}

// ---------------- kernel 2: V [B][N][DK] f32  ->  Vt [B][DK][N] bf16 ----------------
__global__ void __launch_bounds__(256)
vtrans_kernel(const float* __restrict__ V, bf16_t* __restrict__ Vt) {
  __shared__ bf16_t tile[32][33];
  const int bx = blockIdx.x % (DKc / 32);
  const int by = (blockIdx.x / (DKc / 32)) % (Nc / 32);
  const int bz = blockIdx.x / ((DKc / 32) * (Nc / 32));
  const int tx = threadIdx.x & 31, ty = threadIdx.x >> 5;
  const float* src = V + ((size_t)bz * Nc + by * 32) * DKc + bx * 32;
  for (int j = ty; j < 32; j += 8) tile[j][tx] = (bf16_t)src[(size_t)j * DKc + tx];
  __syncthreads();
  bf16_t* dst = Vt + ((size_t)bz * DKc + bx * 32) * Nc + by * 32;
  for (int j = ty; j < 32; j += 8) dst[(size_t)j * Nc + tx] = tile[tx][j];
}

// ---------------- kernel 3: fused flash attention ----------------
// 1 WG = 4 waves = 16 queries; KT=64 keys/iter; wave w: S tile for keys [16w,16w+16),
// PV over its 128 V-columns. TDM double-buffers V^T and pk tiles into LDS.
__global__ void __launch_bounds__(128)
attn_kernel(const bf16_t* __restrict__ pq, const bf16_t* __restrict__ pk,
            const bf16_t* __restrict__ vt, float* __restrict__ out) {
  extern __shared__ char smem[];
  bf16_t* Vbuf = (bf16_t*)smem;                                        // [2][512][VSTR]
  bf16_t* Kbuf = (bf16_t*)(smem + 2 * DKc * VSTR * 2);                 // [2][64][VSTR]
  bf16_t* Pbuf = (bf16_t*)(smem + 2 * DKc * VSTR * 2 + 2 * KT * VSTR * 2); // [16][PSTR]
  float* redmax = (float*)(Pbuf + 16 * PSTR);                          // [4][16]
  float* redsum = redmax + 64;                                         // [4][16]

  const int tid  = threadIdx.x;
  const int wave = tid >> 5, lane = tid & 31;
  const int hi   = lane >> 4, ml = lane & 15;
  const int bidx = blockIdx.x / (Nc / 16);
  const int q0   = (blockIdx.x % (Nc / 16)) * 16;

  const float SL2 = 0.125f * 1.44269504f;   // (1/sqrt(DM)) * log2(e)

  // resident pq A fragments (q rows, K = DM = 64 in two 32-chunks)
  ABv aq0, aq1;
  {
    const bf16_t* pr = pq + ((size_t)bidx * Nc + q0 + ml) * DMc;
    aq0.u[0] = *(const u32x4*)(pr + hi * 8);
    aq0.u[1] = *(const u32x4*)(pr + hi * 8 + 16);
    aq1.u[0] = *(const u32x4*)(pr + 32 + hi * 8);
    aq1.u[1] = *(const u32x4*)(pr + 32 + hi * 8 + 16);
  }

  v8f vzero = {};
  v8f acc[8];
  float rmax[8], rsum[8];
#pragma unroll
  for (int n = 0; n < 8; ++n) acc[n] = vzero;
#pragma unroll
  for (int r = 0; r < 8; ++r) { rmax[r] = -1e30f; rsum[r] = 0.0f; }

  const size_t vbase = (size_t)bidx * DKc * Nc;
  const size_t kbase = (size_t)bidx * Nc * DMc;
  const u32 voff   = (u32)(uintptr_t)(void*)Vbuf;
  const u32 koff   = (u32)(uintptr_t)(void*)Kbuf;
  const u32 VBYTES = DKc * VSTR * 2;
  const u32 KBYTES = KT * VSTR * 2;
  const int T = Nc / KT;

  // preload tile 0
#if HAVE_TDM
  if (wave == 0) {
    tdm_load_2d(voff, vt + vbase, Nc, DKc, Nc, KT, DKc);                 // V^T tile: 64 keys x 512 cols
    tdm_load_2d(koff, pk + kbase, DMc, Nc, DMc, DMc, KT);                // pk tile: 64 keys x 64 dm
  }
#else
  copy_tile_fallback(Vbuf, vt + vbase, DKc, 8, Nc, tid, 128);
  copy_tile_fallback(Kbuf, pk + kbase, KT, 8, DMc, tid, 128);
#endif

  for (int t = 0; t < T; ++t) {
    const int cur = t & 1, nxt = cur ^ 1;
    if (t + 1 < T) {
      const int nk0 = (t + 1) * KT;
#if HAVE_TDM
      if (wave == 0) {
        tdm_load_2d(voff + nxt * VBYTES, vt + vbase + nk0, Nc, DKc, Nc, KT, DKc);
        tdm_load_2d(koff + nxt * KBYTES, pk + kbase + (size_t)nk0 * DMc, DMc, Nc, DMc, DMc, KT);
      }
#else
      copy_tile_fallback(Vbuf + (size_t)nxt * DKc * VSTR, vt + vbase + nk0, DKc, 8, Nc, tid, 128);
      copy_tile_fallback(Kbuf + (size_t)nxt * KT * VSTR, pk + kbase + (size_t)nk0 * DMc, KT, 8, DMc, tid, 128);
#endif
    }
#if HAVE_TDM
    if (wave == 0) {
      if (t + 1 < T) __builtin_amdgcn_s_wait_tensorcnt(2);  // current tile's 2 TDM ops done
      else           __builtin_amdgcn_s_wait_tensorcnt(0);
    }
#endif
    __syncthreads();

    const bf16_t* Vb = Vbuf + (size_t)cur * DKc * VSTR;
    const bf16_t* Kb = Kbuf + (size_t)cur * KT * VSTR;

    // ---- S = pq · pk^T for this wave's 16 keys (B: n=key, k=dm contiguous in pk rows)
    ABv bk0, bk1;
    {
      const bf16_t* kr = Kb + (size_t)(wave * 16 + ml) * VSTR + hi * 8;
      bk0.u[0] = *(const u32x4*)(kr);
      bk0.u[1] = *(const u32x4*)(kr + 16);
      bk1.u[0] = *(const u32x4*)(kr + 32);
      bk1.u[1] = *(const u32x4*)(kr + 48);
    }
    v8f s = vzero;
    s = WMMA_BF16(aq0.v, bk0.v, s);
    s = WMMA_BF16(aq1.v, bk1.v, s);

    // ---- per-tile row maxima (reduce across 16-lane halves) -> LDS
#pragma unroll
    for (int r = 0; r < 8; ++r) {
      float v = s[r];
      v = fmaxf(v, __shfl_xor(v, 1, 32));
      v = fmaxf(v, __shfl_xor(v, 2, 32));
      v = fmaxf(v, __shfl_xor(v, 4, 32));
      v = fmaxf(v, __shfl_xor(v, 8, 32));
      if (ml == 0) redmax[wave * 16 + r + hi * 8] = v;
    }
    __syncthreads();

    // ---- online-softmax running max / rescale factor per owned row
    float fac[8];
#pragma unroll
    for (int r = 0; r < 8; ++r) {
      const int row = r + hi * 8;
      float gm = rmax[r];
      gm = fmaxf(gm, redmax[row]);
      gm = fmaxf(gm, redmax[16 + row]);
      gm = fmaxf(gm, redmax[32 + row]);
      gm = fmaxf(gm, redmax[48 + row]);
      fac[r]  = exp2f((rmax[r] - gm) * SL2);
      rmax[r] = gm;
      rsum[r] *= fac[r];
    }

    // ---- P = exp2((S-m)*scale*log2e): bf16 to LDS (A layout), plus row sums
#pragma unroll
    for (int r = 0; r < 8; ++r) {
      const int row = r + hi * 8;
      float p = exp2f((s[r] - rmax[r]) * SL2);
      Pbuf[row * PSTR + wave * 16 + ml] = (bf16_t)p;
      float sm = p;
      sm += __shfl_xor(sm, 1, 32);
      sm += __shfl_xor(sm, 2, 32);
      sm += __shfl_xor(sm, 4, 32);
      sm += __shfl_xor(sm, 8, 32);
      if (ml == 0) redsum[wave * 16 + row] = sm;
    }
    __syncthreads();

#pragma unroll
    for (int r = 0; r < 8; ++r) {
      const int row = r + hi * 8;
      rsum[r] += redsum[row] + redsum[16 + row] + redsum[32 + row] + redsum[48 + row];
    }

    // ---- rescale accumulators
#pragma unroll
    for (int n = 0; n < 8; ++n)
#pragma unroll
      for (int r = 0; r < 8; ++r) acc[n][r] = acc[n][r] * fac[r];

    // ---- O += P · V : A = P (16q x 64k), B = V^T tile (k=key contiguous per col row)
    ABv pa0, pa1;
    {
      const bf16_t* pr = Pbuf + (size_t)ml * PSTR;
      pa0.u[0] = *(const u32x4*)(pr + hi * 8);
      pa0.u[1] = *(const u32x4*)(pr + hi * 8 + 16);
      pa1.u[0] = *(const u32x4*)(pr + 32 + hi * 8);
      pa1.u[1] = *(const u32x4*)(pr + 32 + hi * 8 + 16);
    }
#pragma unroll
    for (int n = 0; n < 8; ++n) {
      const bf16_t* vr = Vb + (size_t)(wave * 128 + n * 16 + ml) * VSTR + hi * 8;
      ABv vb0, vb1;
      vb0.u[0] = *(const u32x4*)(vr);
      vb0.u[1] = *(const u32x4*)(vr + 16);
      vb1.u[0] = *(const u32x4*)(vr + 32);
      vb1.u[1] = *(const u32x4*)(vr + 48);
      acc[n] = WMMA_BF16(pa0.v, vb0.v, acc[n]);
      acc[n] = WMMA_BF16(pa1.v, vb1.v, acc[n]);
    }
    __syncthreads();   // everyone done reading cur buffers before they get overwritten
  }

  // ---- epilogue: O / l, f32 stores (C layout: M=row, N=col=ml)
  float* op = out + ((size_t)bidx * Nc + q0) * DKc;
#pragma unroll
  for (int r = 0; r < 8; ++r) {
    const int row = r + hi * 8;
    const float inv = 1.0f / rsum[r];
#pragma unroll
    for (int n = 0; n < 8; ++n)
      op[(size_t)row * DKc + wave * 128 + n * 16 + ml] = acc[n][r] * inv;
  }
}

// ---------------- host launcher ----------------
extern "C" void kernel_launch(void* const* d_in, const int* in_sizes, int n_in,
                              void* d_out, int out_size, void* d_ws, size_t ws_size,
                              hipStream_t stream) {
  const float* K  = (const float*)d_in[0];
  const float* Q  = (const float*)d_in[1];
  const float* V  = (const float*)d_in[2];
  const float* Wk = (const float*)d_in[3];
  const float* bk = (const float*)d_in[4];
  const float* Wq = (const float*)d_in[5];
  const float* bq = (const float*)d_in[6];
  float* out = (float*)d_out;

  char* ws = (char*)d_ws;
  bf16_t* pk  = (bf16_t*)(ws);                                    //  4 MB
  bf16_t* pq  = (bf16_t*)(ws + ((size_t)4 << 20));                //  4 MB
  bf16_t* vt  = (bf16_t*)(ws + ((size_t)8 << 20));                // 32 MB
  bf16_t* wkb = (bf16_t*)(ws + ((size_t)40 << 20));               // 64 KB
  bf16_t* wqb = (bf16_t*)(ws + ((size_t)40 << 20) + (1 << 16));   // 64 KB

  const int wn = DMc * DKc;
  cvt_w_kernel<<<(wn + 255) / 256, 256, 0, stream>>>(Wk, wkb, wn);
  cvt_w_kernel<<<(wn + 255) / 256, 256, 0, stream>>>(Wq, wqb, wn);

  proj_kernel<<<(Bc * Nc) / 128, 256, 0, stream>>>(K, wkb, bk, pk);
  proj_kernel<<<(Bc * Nc) / 128, 256, 0, stream>>>(Q, wqb, bq, pq);

  vtrans_kernel<<<Bc * (Nc / 32) * (DKc / 32), 256, 0, stream>>>(V, vt);

  const size_t smem = (size_t)2 * DKc * VSTR * 2   // V^T double buffer
                    + (size_t)2 * KT * VSTR * 2    // pk double buffer
                    + (size_t)16 * PSTR * 2        // P tile
                    + 2 * 64 * sizeof(float);      // reductions  => 168704 B (< 320 KB WGP)
  attn_kernel<<<(Bc * Nc) / 16, 128, smem, stream>>>(pq, pk, vt, out);
}
